// NonLocalAttn_18665927868968
// MI455X (gfx1250) — matrix operand order repaired and checked by reading, not compiled, not run on hardware
//
#include <hip/hip_runtime.h>
#include <hip/hip_bf16.h>
#include <cstddef>

typedef __bf16 bf16_t;
typedef __attribute__((ext_vector_type(16))) __bf16 v16bf;
typedef __attribute__((ext_vector_type(8)))  __bf16 v8bf;
typedef __attribute__((ext_vector_type(8)))  float  v8f;

constexpr int NB = 32;    // batch
constexpr int NC = 512;   // channels
constexpr int NN = 1024;  // H*W
constexpr int NR = 64;    // reduced channels
constexpr float kScale = 0.04419417382415922f; // NC^-0.5
constexpr float kEps = 1e-5f;

// A-operand (16x32 bf16) K index this code uses for element e in lane-half h:
// K = 16h + e (contiguous, same order as the B operand)
__device__ __forceinline__ int kA(int e, int h) {
    return 16 * h + e;
}
// B-operand (32x16 bf16): lanes 0-15 hold K=0..15, lanes 16-31 K=16..31 (one 32B run)

__device__ __forceinline__ v8f wmma_bf16(v16bf a, v16bf b, v8f c) {
    return __builtin_amdgcn_wmma_f32_16x16x32_bf16(false, a, false, b, (short)0, c,
                                                   false, false);
}

__device__ __forceinline__ v8f v8f_zero() {
    v8f z;
#pragma unroll
    for (int i = 0; i < 8; ++i) z[i] = 0.f;
    return z;
}

// Assemble an A-operand register from two aligned 16B halves.
__device__ __forceinline__ v16bf a_from_halves(const bf16_t* p, int h) {
    const v8bf lo = *(const v8bf*)(p + 16 * h);
    const v8bf hi = *(const v8bf*)(p + 16 * h + 8);
    v16bf a;
#pragma unroll
    for (int e = 0; e < 8; ++e) { a[e] = lo[e]; a[e + 8] = hi[e]; }
    return a;
}

// ---------------------------------------------------------------------------
// Kernel 0: one-shot f32 -> bf16 conversion of the four weight matrices
// (each R*C = 32768 elements; 4 * 32768 = 512 blocks * 256 threads exactly).
// ---------------------------------------------------------------------------
__global__ __launch_bounds__(256) void cvt_w_kernel(
    const float* __restrict__ Wt, const float* __restrict__ Wp,
    const float* __restrict__ Wg, const float* __restrict__ Wz,
    bf16_t* __restrict__ WtB, bf16_t* __restrict__ WpB,
    bf16_t* __restrict__ WgB, bf16_t* __restrict__ WzB)
{
    const int idx = blockIdx.x * 256 + threadIdx.x;
    const int seg = idx >> 15;          // 32768 elements per matrix
    const int off = idx & 32767;
    const float* src[4] = {Wt, Wp, Wg, Wz};
    bf16_t*      dst[4] = {WtB, WpB, WgB, WzB};
    dst[seg][off] = (bf16_t)src[seg][off];
}

// ---------------------------------------------------------------------------
// Kernel 1: t/p/g projections.  Per wave: one 16-row n-tile, all R=64 columns,
// all three weight matrices.  x is read exactly once across the grid; weights
// are pre-converted bf16 so the B operand is a single 32B load.
// t,p stored [B,N,R] bf16; g stored [B,R,N] bf16 (transposed for attn A-op).
// ---------------------------------------------------------------------------
__global__ __launch_bounds__(128) void proj_kernel(
    const float* __restrict__ x,
    const bf16_t* __restrict__ WtB, const float* __restrict__ bt,
    const bf16_t* __restrict__ WpB, const float* __restrict__ bp,
    const bf16_t* __restrict__ WgB, const float* __restrict__ bg,
    bf16_t* __restrict__ Tt, bf16_t* __restrict__ Pm, bf16_t* __restrict__ Gt)
{
    const int lane = threadIdx.x & 31;
    const int wv   = threadIdx.x >> 5;
    const int tile = blockIdx.x * 4 + wv;       // NB * (NN/16) tiles
    const int b    = tile >> 6;
    const int n0   = (tile & 63) << 4;
    const int l = lane & 15, h = lane >> 4;

    v8f acc[3][4];
#pragma unroll
    for (int w = 0; w < 3; ++w)
#pragma unroll
        for (int rt = 0; rt < 4; ++rt) acc[w][rt] = v8f_zero();

    const bf16_t* Ws[3] = {WtB, WpB, WgB};
    const float* xbase = x + (size_t)b * NC * NN + n0 + l;   // x[b, c, n0+l]

    for (int kc = 0; kc < NC / 32; ++kc) {
        const int c0 = kc * 32;
        // A operand: x^T tile, rows n, K = c (strided gather, lanes coalesce on n)
        v16bf a;
#pragma unroll
        for (int e = 0; e < 16; ++e)
            a[e] = (bf16_t)xbase[(size_t)(c0 + kA(e, h)) * NN];
#pragma unroll
        for (int w = 0; w < 3; ++w) {
            const bf16_t* wp = Ws[w] + (size_t)l * NC + c0 + 16 * h;
#pragma unroll
            for (int rt = 0; rt < 4; ++rt) {
                // B: row r = rt*16 + l, K contiguous -> one aligned 32B load
                const v16bf bm = *(const v16bf*)(wp + (size_t)rt * 16 * NC);
                acc[w][rt] = wmma_bf16(a, bm, acc[w][rt]);
            }
        }
    }

    const float* biases[3] = {bt, bp, bg};
#pragma unroll
    for (int w = 0; w < 3; ++w) {
#pragma unroll
        for (int rt = 0; rt < 4; ++rt) {
            const int r = rt * 16 + l;
            const float bv = biases[w][r];
            if (w < 2) {
                // D layout: row n = n0 + j + 8h, col r.  Store [B,N,R].
                bf16_t* dst = (w == 0 ? Tt : Pm) +
                              ((size_t)(b * NN + n0 + 8 * h)) * NR + r;
#pragma unroll
                for (int j = 0; j < 8; ++j)
                    dst[(size_t)j * NR] = (bf16_t)(acc[w][rt][j] + bv);
            } else {
                // g transposed: [B,R,N], n contiguous -> one 16B store
                v8bf gv;
#pragma unroll
                for (int j = 0; j < 8; ++j) gv[j] = (bf16_t)(acc[2][rt][j] + bv);
                *(v8bf*)(Gt + ((size_t)(b * NR + r)) * NN + n0 + 8 * h) = gv;
            }
        }
    }
}

// ---------------------------------------------------------------------------
// Kernel 2: flash attention.  Per wave: one 16-wide n-tile; loops m in chunks
// of 32.  Logits computed as att^T (M=m, N=n) so softmax stats per column n
// are in-lane; probs re-packed into the B-operand layout via 16 shfl_xor(16).
// y accumulated as y^T[r,n] in f32; output Y stored [B,N,R] bf16.
// ---------------------------------------------------------------------------
__global__ __launch_bounds__(128) void attn_kernel(
    const bf16_t* __restrict__ Tt, const bf16_t* __restrict__ Pm,
    const bf16_t* __restrict__ Gt, bf16_t* __restrict__ Y)
{
    const int lane = threadIdx.x & 31;
    const int wv   = threadIdx.x >> 5;
    const int tile = blockIdx.x * 4 + wv;
    const int b    = tile >> 6;
    const int n0   = (tile & 63) << 4;
    const int l = lane & 15, h = lane >> 4;

    // Resident B operand: t tile, col n = n0+l, K = r (contiguous 32B/lane)
    const bf16_t* tp = Tt + ((size_t)(b * NN + n0 + l)) * NR + 16 * h;
    const v16bf tB0 = *(const v16bf*)(tp);
    const v16bf tB1 = *(const v16bf*)(tp + 32);

    v8f yacc[4];
#pragma unroll
    for (int rt = 0; rt < 4; ++rt) yacc[rt] = v8f_zero();
    float mrun = -1e30f, srun = 0.f;

    for (int mc = 0; mc < NN / 32; ++mc) {
        const int m0 = mc * 32;
        // A operands: p rows m0..m0+31, K = r (contiguous, 16B-aligned halves)
        const bf16_t* pp0 = Pm + ((size_t)(b * NN + m0 + l)) * NR;
        const bf16_t* pp1 = pp0 + 16 * NR;
        const v16bf a00 = a_from_halves(pp0, h);
        const v16bf a01 = a_from_halves(pp0 + 32, h);
        const v16bf a10 = a_from_halves(pp1, h);
        const v16bf a11 = a_from_halves(pp1 + 32, h);

        v8f d0 = v8f_zero(), d1 = v8f_zero();
        d0 = wmma_bf16(a00, tB0, d0); d0 = wmma_bf16(a01, tB1, d0);
        d1 = wmma_bf16(a10, tB0, d1); d1 = wmma_bf16(a11, tB1, d1);

        // online softmax over m (column n fixed per lane)
        float e0[8], e1[8];
        float cmax = -1e30f;
#pragma unroll
        for (int j = 0; j < 8; ++j) {
            e0[j] = d0[j] * kScale; e1[j] = d1[j] * kScale;
            cmax = fmaxf(cmax, fmaxf(e0[j], e1[j]));
        }
        cmax = fmaxf(cmax, __shfl_xor(cmax, 16, 32));
        const float mnew = fmaxf(mrun, cmax);
        const float corr = __expf(mrun - mnew);
        float part = 0.f;
#pragma unroll
        for (int j = 0; j < 8; ++j) {
            e0[j] = __expf(e0[j] - mnew);
            e1[j] = __expf(e1[j] - mnew);
            part += e0[j] + e1[j];
        }
        srun = srun * corr + part;
        mrun = mnew;
#pragma unroll
        for (int rt = 0; rt < 4; ++rt)
#pragma unroll
            for (int j = 0; j < 8; ++j) yacc[rt][j] *= corr;

        // pack probs into B-operand layout: element e needs m_local = 16h + e
        v16bf pb;
#pragma unroll
        for (int j = 0; j < 8; ++j) {
            const float q0 = __shfl_xor(e0[j], 16, 32);
            const float q1 = __shfl_xor(e1[j], 16, 32);
            pb[j]     = (bf16_t)(h ? q1 : e0[j]);
            pb[j + 8] = (bf16_t)(h ? e1[j] : q0);
        }
        // y^T[r,n] += g^T[r,m] * probs[m,n]
#pragma unroll
        for (int rt = 0; rt < 4; ++rt) {
            const bf16_t* gp = Gt + ((size_t)(b * NR + rt * 16 + l)) * NN + m0;
            const v16bf ga = a_from_halves(gp, h);
            yacc[rt] = wmma_bf16(ga, pb, yacc[rt]);
        }
    }

    const float stot = srun + __shfl_xor(srun, 16, 32);
    const float inv  = 1.f / stot;
    // D layout: r = rt*16 + j + 8h, n = n0 + l  ->  Y[B,N,R], 16B stores
#pragma unroll
    for (int rt = 0; rt < 4; ++rt) {
        v8bf yv;
#pragma unroll
        for (int j = 0; j < 8; ++j) yv[j] = (bf16_t)(yacc[rt][j] * inv);
        *(v8bf*)(Y + ((size_t)(b * NN + n0 + l)) * NR + rt * 16 + 8 * h) = yv;
    }
}

// ---------------------------------------------------------------------------
// Kernel 3: z = Wz * y^T, fused BN (inference) + residual.
// Per wave: 16 channels x 64 positions.  Wz pre-converted to bf16.
// ---------------------------------------------------------------------------
__global__ __launch_bounds__(128) void out_kernel(
    const bf16_t* __restrict__ Y, const bf16_t* __restrict__ WzB,
    const float* __restrict__ bz, const float* __restrict__ gamma,
    const float* __restrict__ beta, const float* __restrict__ bnmean,
    const float* __restrict__ bnvar, const float* __restrict__ x,
    float* __restrict__ out)
{
    const int lane = threadIdx.x & 31;
    const int wv   = threadIdx.x >> 5;
    const int tile = blockIdx.x * 4 + wv;   // NB * (NC/16) * (NN/64)
    const int b    = tile >> 9;
    const int ct   = (tile >> 4) & 31;
    const int ng   = tile & 15;
    const int c0 = ct * 16, n0 = ng * 64;
    const int l = lane & 15, h = lane >> 4;

    // Resident A operand: Wz rows c = c0 + l, K = r (contiguous, 16B halves)
    const bf16_t* wzp = WzB + (size_t)(c0 + l) * NR;
    const v16bf wa0 = a_from_halves(wzp, h);
    const v16bf wa1 = a_from_halves(wzp + 32, h);

    // Fold bias + BN into one scale/shift per (lane, j): out = z*sc + sh + x
    float sc[8], sh[8];
#pragma unroll
    for (int j = 0; j < 8; ++j) {
        const int c = c0 + 8 * h + j;
        const float iv = rsqrtf(bnvar[c] + kEps);
        const float g  = gamma[c] * iv;
        sc[j] = g;
        sh[j] = (bz[c] - bnmean[c]) * g + beta[c];
    }

#pragma unroll
    for (int nt = 0; nt < 4; ++nt) {
        const int n1 = n0 + nt * 16;
        const bf16_t* yp = Y + ((size_t)(b * NN + n1 + l)) * NR + 16 * h;
        const v16bf yB0 = *(const v16bf*)(yp);
        const v16bf yB1 = *(const v16bf*)(yp + 32);
        v8f accz = v8f_zero();
        accz = wmma_bf16(wa0, yB0, accz);
        accz = wmma_bf16(wa1, yB1, accz);

        const size_t base = ((size_t)b * NC + c0 + 8 * h) * NN + n1 + l;
#pragma unroll
        for (int j = 0; j < 8; ++j)
            out[base + (size_t)j * NN] =
                accz[j] * sc[j] + sh[j] + x[base + (size_t)j * NN];
    }
}

// ---------------------------------------------------------------------------
extern "C" void kernel_launch(void* const* d_in, const int* in_sizes, int n_in,
                              void* d_out, int out_size, void* d_ws, size_t ws_size,
                              hipStream_t stream) {
    const float* x     = (const float*)d_in[0];
    const float* Wt    = (const float*)d_in[1];
    const float* bt    = (const float*)d_in[2];
    const float* Wp    = (const float*)d_in[3];
    const float* bp    = (const float*)d_in[4];
    const float* Wg    = (const float*)d_in[5];
    const float* bg    = (const float*)d_in[6];
    const float* Wz    = (const float*)d_in[7];
    const float* bz    = (const float*)d_in[8];
    const float* gamma = (const float*)d_in[9];
    const float* beta  = (const float*)d_in[10];
    const float* bmean = (const float*)d_in[11];
    const float* bvar  = (const float*)d_in[12];
    float* out = (float*)d_out;

    // Workspace: 4 bf16 intermediates of B*N*R = 4 MB each, then bf16 weights.
    char* ws = (char*)d_ws;
    bf16_t* Tt  = (bf16_t*)(ws);
    bf16_t* Pm  = (bf16_t*)(ws + (size_t)4 * 1024 * 1024);
    bf16_t* Gt  = (bf16_t*)(ws + (size_t)8 * 1024 * 1024);
    bf16_t* Yb  = (bf16_t*)(ws + (size_t)12 * 1024 * 1024);
    bf16_t* WtB = (bf16_t*)(ws + (size_t)16 * 1024 * 1024);
    bf16_t* WpB = WtB + (size_t)NR * NC;
    bf16_t* WgB = WpB + (size_t)NR * NC;
    bf16_t* WzB = WgB + (size_t)NR * NC;

    // One-shot weight conversion: 4 * 32768 elements = 512 * 256 threads.
    cvt_w_kernel<<<512, 256, 0, stream>>>(Wt, Wp, Wg, Wz, WtB, WpB, WgB, WzB);
    // 2048 waves (NB * NN/16), 4 waves / 128-thread block
    proj_kernel<<<512, 128, 0, stream>>>(x, WtB, bt, WpB, bp, WgB, bg, Tt, Pm, Gt);
    attn_kernel<<<512, 128, 0, stream>>>(Tt, Pm, Gt, Yb);
    // 16384 waves (NB * NC/16 * NN/64)
    out_kernel<<<4096, 128, 0, stream>>>(Yb, WzB, bz, gamma, beta, bmean, bvar, x, out);
}
